// MambaDecoder_67113158967799
// MI455X (gfx1250) — matrix operand order repaired and checked by reading, not compiled, not run on hardware
//
#include <hip/hip_runtime.h>

// ---------------- CDNA5 WMMA types ----------------
typedef __attribute__((ext_vector_type(16))) __bf16    v16bf;
typedef __attribute__((ext_vector_type(8)))  float     v8f;
typedef __attribute__((ext_vector_type(4)))  unsigned  uint4v;

union Frag { v16bf v; uint4v q[2]; };   // 32B: one WMMA A/B fragment per lane

#define DEV __device__ __forceinline__

DEV unsigned short f2bf(float f) {
  union { float f; unsigned u; } x; x.f = f;
  unsigned u = x.u;
  unsigned r = (u + 0x7FFFu + ((u >> 16) & 1u)) >> 16;   // round-nearest-even
  return (unsigned short)r;
}

DEV float sigmoidf_(float x) { return 1.0f / (1.0f + __expf(-x)); }

// ---------------- constants ----------------
#define D_MODEL 768
#define D_INNER 1536
#define D_STATE 16
#define D_CONV  4
#define DT_RANK 48
#define VOCAB   50257
#define BATCH   2
#define SEQL    1024
#define MROWS   (BATCH * SEQL)          // 2048

// =====================================================================
// Embedding gather: h[row, :] = embed[ids[row], :]
// =====================================================================
__global__ void k_embed(const int* __restrict__ ids,
                        const float* __restrict__ embed,
                        float* __restrict__ h) {
  int row = blockIdx.x;
  int id  = ids[row];
  const float* src = embed + (size_t)id * D_MODEL;
  float* dst = h + (size_t)row * D_MODEL;
  for (int c = threadIdx.x; c < D_MODEL; c += blockDim.x) dst[c] = src[c];
}

// =====================================================================
// RMSNorm row-wise, fp32 in -> bf16 out (ready as WMMA A operand)
// =====================================================================
__global__ __launch_bounds__(256)
void k_rmsnorm_bf16(const float* __restrict__ x, const float* __restrict__ w,
                    unsigned short* __restrict__ out, int D) {
  int row = blockIdx.x;
  const float* xr = x + (size_t)row * D;
  float s = 0.f;
  for (int c = threadIdx.x; c < D; c += 256) { float v = xr[c]; s += v * v; }
  #pragma unroll
  for (int o = 16; o > 0; o >>= 1) s += __shfl_xor(s, o, 32);   // wave32 reduce
  __shared__ float red[8];
  if ((threadIdx.x & 31) == 0) red[threadIdx.x >> 5] = s;
  __syncthreads();
  float tot = 0.f;
  #pragma unroll
  for (int i = 0; i < 8; ++i) tot += red[i];
  float rs = rsqrtf(tot / (float)D + 1e-5f);
  for (int c = threadIdx.x; c < D; c += 256)
    out[(size_t)row * D + c] = f2bf(xr[c] * rs * w[c]);
}

// =====================================================================
// Generic fp32 -> bf16 convert with column slice + zero pad.
// =====================================================================
__global__ void k_cvt(const float* __restrict__ src, unsigned short* __restrict__ dst,
                      int rows, int sstride, int coff, int copyc, int dstride) {
  long long n = (long long)rows * dstride;
  long long i = (long long)blockIdx.x * blockDim.x + threadIdx.x;
  if (i >= n) return;
  int r = (int)(i / dstride), c = (int)(i % dstride);
  dst[i] = (c < copyc) ? f2bf(src[(size_t)r * sstride + coff + c]) : (unsigned short)0;
}

// =====================================================================
// bf16 WMMA GEMM:  C[M,N] (+)= A[M,K] * Bw[N,K]^T   (row-major)
//
// Block 256 thr = 8 waves (2 M x 4 N); block tile 64x128; wave tile 32x32
// (4 accumulators). LDS tiles stored FRAGMENT-NATIVE: each lane's 16
// bf16 A/B fragment is 32 contiguous bytes -> 2x ds_load_b128 per frag,
// 4 v_wmma per 8 ds_load_b128. Double-buffered LDS, 1 barrier / K-step,
// global_prefetch_b8 for the K+2 tile.
//
// Fragment-native addressing (contiguous k order k = 16*half + j, same for A and B):
//   A elem (m,k): lane=(m&15)+16*(k>>4),     slot j=k&15
//   B elem (n,k): lane=(n&15)+16*(k>>4),     slot j=k&15
// => every aligned 8-wide k-group is one 16B LDS slot.
// =====================================================================
__global__ __launch_bounds__(256)
void k_gemm_bf16(const unsigned short* __restrict__ A,
                 const unsigned short* __restrict__ Bw,
                 float* __restrict__ C,
                 int M, int N, int K, int acc) {
  __shared__ unsigned short lA[2][64 * 32];    // 4KB per buffer, frag-native
  __shared__ unsigned short lB[2][128 * 32];   // 8KB per buffer, frag-native

  const int tid  = threadIdx.x;
  const int lane = tid & 31;
  const int wave = tid >> 5;
  const int wm   = wave & 1;            // 0..1 : 32-row slab
  const int wn   = wave >> 1;           // 0..3 : 32-col slab
  const int half = lane >> 4;
  const int mr   = lane & 15;
  const int mBlock = blockIdx.y * 64;
  const int nBlock = blockIdx.x * 128;

  // ---- A loader: thread -> (row am = tid>>2, k-group ag = tid&3) ----
  const int am = tid >> 2;
  const int ag = tid & 3;
  const int aRow = mBlock + am;
  const long long aOff = (long long)aRow * K + ag * 8;
  const int aLds = (am >> 4) * 512 + ((am & 15) + 16 * (ag >> 1)) * 16 + (ag & 1) * 8;

  // ---- B loader: thread -> (col bn = tid>>1, k-groups {2t,2t+1}, t = tid&1) ----
  const int bn = tid >> 1;
  const int bt = tid & 1;
  const int bCol = nBlock + bn;
  const long long bOff = (long long)bCol * K + bt * 16;
  const int bLds = (bn >> 4) * 512 + ((bn & 15) + 16 * bt) * 16;

  const int nk = K >> 5;

  auto loadTile = [&](int ki, int buf) {
    const int k0 = ki << 5;
    uint4v av = {0u, 0u, 0u, 0u};
    if (aRow < M) {
      av = *(const uint4v*)(A + aOff + k0);
      if (k0 + 64 < K) __builtin_prefetch(A + aOff + k0 + 64, 0, 1);
    }
    *(uint4v*)&lA[buf][aLds] = av;

    uint4v b0 = {0u, 0u, 0u, 0u}, b1 = {0u, 0u, 0u, 0u};
    if (bCol < N) {
      const uint4v* p = (const uint4v*)(Bw + bOff + k0);
      b0 = p[0]; b1 = p[1];
      if (k0 + 64 < K) __builtin_prefetch(Bw + bOff + k0 + 64, 0, 1);
    }
    uint4v* q = (uint4v*)&lB[buf][bLds];
    q[0] = b0; q[1] = b1;
  };

  const v8f vzero = {0.f, 0.f, 0.f, 0.f, 0.f, 0.f, 0.f, 0.f};
  v8f accv[2][2];
  #pragma unroll
  for (int mi = 0; mi < 2; ++mi)
    #pragma unroll
    for (int ni = 0; ni < 2; ++ni) accv[mi][ni] = vzero;

  if (acc) {                                   // C += ...
    #pragma unroll
    for (int mi = 0; mi < 2; ++mi)
      #pragma unroll
      for (int ni = 0; ni < 2; ++ni) {
        int col = nBlock + wn * 32 + ni * 16 + mr;
        if (col < N) {
          #pragma unroll
          for (int v = 0; v < 8; ++v) {
            int row = mBlock + wm * 32 + mi * 16 + v + 8 * half;
            if (row < M) accv[mi][ni][v] = C[(size_t)row * N + col];
          }
        }
      }
  }

  loadTile(0, 0);
  for (int ki = 0; ki < nk; ++ki) {
    __syncthreads();
    if (ki + 1 < nk) loadTile(ki + 1, (ki + 1) & 1);

    const int buf = ki & 1;
    Frag afr[2], bfr[2];
    #pragma unroll
    for (int mi = 0; mi < 2; ++mi) {
      const uint4v* p = (const uint4v*)&lA[buf][(wm * 2 + mi) * 512 + lane * 16];
      afr[mi].q[0] = p[0]; afr[mi].q[1] = p[1];
    }
    #pragma unroll
    for (int ni = 0; ni < 2; ++ni) {
      const uint4v* p = (const uint4v*)&lB[buf][(wn * 2 + ni) * 512 + lane * 16];
      bfr[ni].q[0] = p[0]; bfr[ni].q[1] = p[1];
    }
    #pragma unroll
    for (int mi = 0; mi < 2; ++mi)
      #pragma unroll
      for (int ni = 0; ni < 2; ++ni)
        accv[mi][ni] = __builtin_amdgcn_wmma_f32_16x16x32_bf16(
            false, afr[mi].v, false, bfr[ni].v,
            (short)0, accv[mi][ni], false, false);
  }

  #pragma unroll
  for (int mi = 0; mi < 2; ++mi)
    #pragma unroll
    for (int ni = 0; ni < 2; ++ni) {
      int col = nBlock + wn * 32 + ni * 16 + mr;
      if (col < N) {
        #pragma unroll
        for (int v = 0; v < 8; ++v) {
          int row = mBlock + wm * 32 + mi * 16 + v + 8 * half;
          if (row < M) C[(size_t)row * N + col] = accv[mi][ni][v];
        }
      }
    }
}

// =====================================================================
// Causal depthwise conv (width 4) + bias + SiLU from xz[:, :D_INNER].
// =====================================================================
__global__ void k_conv_silu(const float* __restrict__ xz,
                            const float* __restrict__ cw, const float* __restrict__ cb,
                            float* __restrict__ xcf, unsigned short* __restrict__ xcbf) {
  int i = blockIdx.x * blockDim.x + threadIdx.x;
  if (i >= BATCH * SEQL * D_INNER) return;
  int d = i % D_INNER;
  int l = (i / D_INNER) % SEQL;
  int b = i / (D_INNER * SEQL);
  const float* col = xz + (size_t)b * SEQL * (2 * D_INNER) + d;
  float acc = cb[d];
  #pragma unroll
  for (int j = 0; j < D_CONV; ++j) {
    int ls = l - (D_CONV - 1) + j;
    if (ls >= 0) acc += cw[d * D_CONV + j] * col[(size_t)ls * (2 * D_INNER)];
  }
  float sv = acc * sigmoidf_(acc);
  xcf[i]  = sv;
  xcbf[i] = f2bf(sv);
}

// =====================================================================
// dt = softplus(dt_raw + bias)   (in place)
// =====================================================================
__global__ void k_bias_softplus(float* __restrict__ dt, const float* __restrict__ b) {
  int i = blockIdx.x * blockDim.x + threadIdx.x;
  if (i >= MROWS * D_INNER) return;
  float x = dt[i] + b[i % D_INNER];
  dt[i] = (x > 20.f) ? x : log1pf(__expf(x));
}

// =====================================================================
// Selective scan. One thread per (b, channel); 16 fp32 states in regs.
// B/C vectors staged 8 timesteps per LDS round (2 barriers / 8 steps).
// =====================================================================
__global__ __launch_bounds__(256)
void k_scan(const float* __restrict__ xcf, const float* __restrict__ dt,
            const float* __restrict__ dbc, const float* __restrict__ A_log,
            const float* __restrict__ Dsk, const float* __restrict__ xz,
            unsigned short* __restrict__ ybf) {
  const int b = blockIdx.x / (D_INNER / 256);
  const int d = (blockIdx.x % (D_INNER / 256)) * 256 + threadIdx.x;
  float Ar[D_STATE], st[D_STATE];
  #pragma unroll
  for (int s = 0; s < D_STATE; ++s) {
    Ar[s] = -__expf(A_log[(size_t)d * D_STATE + s]);
    st[s] = 0.f;
  }
  const float Dv = Dsk[d];
  __shared__ float sBC[8][2 * D_STATE];      // [timestep][B(16) | C(16)]
  const int lrow = threadIdx.x >> 5;         // 0..7
  const int lcol = threadIdx.x & 31;         // 0..31
  for (int l0 = 0; l0 < SEQL; l0 += 8) {
    __syncthreads();
    {
      size_t row = (size_t)b * SEQL + l0 + lrow;
      sBC[lrow][lcol] = dbc[row * (DT_RANK + 2 * D_STATE) + DT_RANK + lcol];
    }
    __syncthreads();
    #pragma unroll
    for (int j = 0; j < 8; ++j) {
      size_t row = (size_t)b * SEQL + l0 + j;
      float dtv = dt[row * D_INNER + d];
      float xv  = xcf[row * D_INNER + d];
      float y = 0.f;
      #pragma unroll
      for (int s = 0; s < D_STATE; ++s) {
        float dA = __expf(dtv * Ar[s]);
        st[s] = dA * st[s] + dtv * sBC[j][s] * xv;
        y += st[s] * sBC[j][D_STATE + s];
      }
      y += xv * Dv;
      float zv = xz[row * (2 * D_INNER) + D_INNER + d];
      ybf[row * D_INNER + d] = f2bf(y * (zv * sigmoidf_(zv)));
    }
  }
}

// =====================================================================
// Host orchestration
// =====================================================================
extern "C" void kernel_launch(void* const* d_in, const int* in_sizes, int n_in,
                              void* d_out, int out_size, void* d_ws, size_t ws_size,
                              hipStream_t stream) {
  const int*   ids    = (const int*)  d_in[0];
  const float* embed  = (const float*)d_in[1];
  const float* in_w   = (const float*)d_in[2];
  const float* conv_w = (const float*)d_in[3];
  const float* conv_b = (const float*)d_in[4];
  const float* xp_w   = (const float*)d_in[5];
  const float* dtw    = (const float*)d_in[6];
  const float* dtb    = (const float*)d_in[7];
  const float* A_log  = (const float*)d_in[8];
  const float* Dsk    = (const float*)d_in[9];
  const float* out_w  = (const float*)d_in[10];
  const float* norm_w = (const float*)d_in[11];
  const float* norm_f = (const float*)d_in[12];
  float* out = (float*)d_out;

  // ---- carve workspace (256B aligned) ----
  char* p = (char*)d_ws;
  auto alloc = [&](size_t bytes) -> void* {
    void* r = (void*)p;
    p += (bytes + 255) & ~(size_t)255;
    return r;
  };
  float*          h     = (float*)         alloc((size_t)MROWS * D_MODEL * 4);
  unsigned short* xbf   = (unsigned short*)alloc((size_t)MROWS * D_MODEL * 2);
  float*          xz    = (float*)         alloc((size_t)MROWS * 2 * D_INNER * 4);
  float*          xcf   = (float*)         alloc((size_t)MROWS * D_INNER * 4);
  unsigned short* xcbf  = (unsigned short*)alloc((size_t)MROWS * D_INNER * 2);
  float*          dbcB  = (float*)         alloc((size_t)MROWS * (DT_RANK + 2 * D_STATE) * 4);
  unsigned short* dtin  = (unsigned short*)alloc((size_t)MROWS * 64 * 2);
  float*          dtbuf = (float*)         alloc((size_t)MROWS * D_INNER * 4);
  unsigned short* ybf   = (unsigned short*)alloc((size_t)MROWS * D_INNER * 2);
  unsigned short* wbuf  = (unsigned short*)alloc((size_t)2 * D_INNER * D_MODEL * 2);
  unsigned short* embbf = (unsigned short*)alloc((size_t)VOCAB * D_MODEL * 2);

  auto blks = [](long long n) { return (unsigned)((n + 255) / 256); };

  k_embed<<<MROWS, 256, 0, stream>>>(ids, embed, h);

  for (int i = 0; i < 4; ++i) {
    // x = rmsnorm(h) -> bf16
    k_rmsnorm_bf16<<<MROWS, 256, 0, stream>>>(h, norm_w + (size_t)i * D_MODEL, xbf, D_MODEL);

    // xz = x @ in_proj_w^T       (2048 x 3072, K=768)
    long long nw = (long long)2 * D_INNER * D_MODEL;
    k_cvt<<<blks(nw), 256, 0, stream>>>(in_w + (size_t)i * 2 * D_INNER * D_MODEL, wbuf,
                                        2 * D_INNER, D_MODEL, 0, D_MODEL, D_MODEL);
    dim3 g1((2 * D_INNER + 127) / 128, MROWS / 64);
    k_gemm_bf16<<<g1, 256, 0, stream>>>(xbf, wbuf, xz, MROWS, 2 * D_INNER, D_MODEL, 0);

    // xc = silu(conv(xz[:, :D_INNER]))
    k_conv_silu<<<blks((long long)MROWS * D_INNER), 256, 0, stream>>>(
        xz, conv_w + (size_t)i * D_INNER * D_CONV, conv_b + (size_t)i * D_INNER, xcf, xcbf);

    // dbc = xc @ x_proj_w^T      (2048 x 80, K=1536)
    nw = (long long)(DT_RANK + 2 * D_STATE) * D_INNER;
    k_cvt<<<blks(nw), 256, 0, stream>>>(xp_w + (size_t)i * (DT_RANK + 2 * D_STATE) * D_INNER,
                                        wbuf, DT_RANK + 2 * D_STATE, D_INNER, 0, D_INNER, D_INNER);
    dim3 g2((DT_RANK + 2 * D_STATE + 127) / 128, MROWS / 64);
    k_gemm_bf16<<<g2, 256, 0, stream>>>(xcbf, wbuf, dbcB, MROWS, DT_RANK + 2 * D_STATE, D_INNER, 0);

    // dt_raw = dbc[:, :48] @ dt_proj_w^T   (K padded 48 -> 64 with zeros)
    k_cvt<<<blks((long long)MROWS * 64), 256, 0, stream>>>(dbcB, dtin, MROWS,
                                                           DT_RANK + 2 * D_STATE, 0, DT_RANK, 64);
    k_cvt<<<blks((long long)D_INNER * 64), 256, 0, stream>>>(dtw + (size_t)i * D_INNER * DT_RANK,
                                                             wbuf, D_INNER, DT_RANK, 0, DT_RANK, 64);
    dim3 g3((D_INNER + 127) / 128, MROWS / 64);
    k_gemm_bf16<<<g3, 256, 0, stream>>>(dtin, wbuf, dtbuf, MROWS, D_INNER, 64, 0);
    k_bias_softplus<<<blks((long long)MROWS * D_INNER), 256, 0, stream>>>(
        dtbuf, dtb + (size_t)i * D_INNER);

    // selective scan (+ D skip, * silu(z)) -> bf16 y
    k_scan<<<BATCH * (D_INNER / 256), 256, 0, stream>>>(
        xcf, dtbuf, dbcB, A_log + (size_t)i * D_INNER * D_STATE,
        Dsk + (size_t)i * D_INNER, xz, ybf);

    // h += y @ out_proj_w^T      (2048 x 768, K=1536, accumulate)
    nw = (long long)D_MODEL * D_INNER;
    k_cvt<<<blks(nw), 256, 0, stream>>>(out_w + (size_t)i * D_MODEL * D_INNER, wbuf,
                                        D_MODEL, D_INNER, 0, D_INNER, D_INNER);
    dim3 g4((D_MODEL + 127) / 128, MROWS / 64);
    k_gemm_bf16<<<g4, 256, 0, stream>>>(ybf, wbuf, h, MROWS, D_MODEL, D_INNER, 1);
  }

  // final norm + logits = hf @ embed^T   (2048 x 50257, K=768)
  k_rmsnorm_bf16<<<MROWS, 256, 0, stream>>>(h, norm_f, xbf, D_MODEL);
  k_cvt<<<blks((long long)VOCAB * D_MODEL), 256, 0, stream>>>(embed, embbf, VOCAB,
                                                              D_MODEL, 0, D_MODEL, D_MODEL);
  dim3 g5((VOCAB + 127) / 128, MROWS / 64);
  k_gemm_bf16<<<g5, 256, 0, stream>>>(xbf, embbf, out, MROWS, VOCAB, D_MODEL, 0);
}
